// Critic_GRU_13726715478472
// MI455X (gfx1250) — matrix operand order repaired and checked by reading, not compiled, not run on hardware
//
#include <hip/hip_runtime.h>
#include <hip/hip_bf16.h>

// ---------------- problem constants ----------------
#define Bm   128
#define Sm   1024
#define BS   (Bm * Sm)      // 131072 positions
#define Em   64
#define NHm  4
#define HDm  16
#define Hm   128
#define LN_EPS 1e-5f

typedef __attribute__((ext_vector_type(16))) _Float16 v16h;
typedef __attribute__((ext_vector_type(8)))  _Float16 v8h;
typedef __attribute__((ext_vector_type(8)))  float    v8f;

__device__ __forceinline__ float leaky(float x) { return x >= 0.f ? x : 0.01f * x; }

// ---------------- WMMA helpers (CDNA5 16x16x32 f16, f32 acc) ----------------
__device__ __forceinline__ v8f wmma16(v16h a, v16h b, v8f c) {
  // 8 args: (neg_a, A, neg_b, B, c_mod, C, reuse_a, reuse_b)
  return __builtin_amdgcn_wmma_f32_16x16x32_f16(false, a, false, b, (short)0, c, false, false);
}

// A matrix 16x32 (M x K), f16. Lane L: row = L%16.
// lanes 0-15 hold K = [0..15]; lanes 16-31 hold K = [16..31].
__device__ __forceinline__ v16h load_tileA(const _Float16* base, int ld, int kBase, int lane) {
  int row = lane & 15;
  int k0  = kBase + ((lane & 16) ? 16 : 0);
  const _Float16* p = base + row * ld + k0;
  union { v16h v; v8h h[2]; } u;
  u.h[0] = *(const v8h*)(p);
  u.h[1] = *(const v8h*)(p + 8);
  return u.v;
}

// B matrix 32x16 (K x N) sourced from row-major W (N x K): B[k][n] = W[nBase+n][kBase+k].
// Lane L: col n = L%16; lanes 0-15 hold K=[0..15], lanes 16-31 hold K=[16..31].
__device__ __forceinline__ v16h load_tileB(const _Float16* W, int ldw, int nBase, int kBase, int lane) {
  int n  = lane & 15;
  int k0 = kBase + ((lane & 16) ? 16 : 0);
  const _Float16* p = W + (nBase + n) * ldw + k0;
  union { v16h v; v8h h[2]; } u;
  u.h[0] = *(const v8h*)(p);
  u.h[1] = *(const v8h*)(p + 8);
  return u.v;
}

// Same as load_tileB but reads f32 weights straight from global and converts,
// for one-time hoisting of B tiles into registers (no LDS staging needed).
__device__ __forceinline__ v16h load_tileB_f32(const float* W, int ldw, int nBase, int kBase, int lane) {
  int n  = lane & 15;
  int k0 = kBase + ((lane & 16) ? 16 : 0);
  const float* p = W + (nBase + n) * ldw + k0;
  v16h v;
  #pragma unroll
  for (int i = 0; i < 16; ++i) v[i] = (_Float16)p[i];
  return v;
}

// C/D f32 16x16: lane L holds col = L%16, rows i + (L<16 ? 0 : 8), i = 0..7.

// ============================================================================
// Kernel 1: encoders + 2-key multi-head attention + out_proj + residual + LN
// One thread per (b, s) position. All weights staged in LDS. Writes xn (f16).
// ============================================================================
__global__ __launch_bounds__(256) void k1_encode_attn_ln(
    const float* __restrict__ obs,
    const float* __restrict__ me_w1, const float* __restrict__ me_b1,
    const float* __restrict__ me_w2, const float* __restrict__ me_b2,
    const float* __restrict__ ae_w1, const float* __restrict__ ae_b1,
    const float* __restrict__ ae_w2, const float* __restrict__ ae_b2,
    const float* __restrict__ in_w,  const float* __restrict__ in_b,
    const float* __restrict__ ow,    const float* __restrict__ ob,
    const float* __restrict__ ln_g,  const float* __restrict__ ln_b,
    _Float16* __restrict__ xn_out)
{
  __shared__ float s_mw1[32*3],  s_mb1[32], s_mw2[64*32], s_mb2[64];
  __shared__ float s_aw1[32*6],  s_ab1[32], s_aw2[64*32], s_ab2[64];
  __shared__ float s_inw[192*64], s_inb[192];
  __shared__ float s_ow[64*64],  s_ob[64];
  __shared__ float s_g[64], s_bb[64];

  const int tid = threadIdx.x;
  for (int i = tid; i < 32*3;   i += 256) s_mw1[i] = me_w1[i];
  for (int i = tid; i < 32;     i += 256) { s_mb1[i] = me_b1[i]; s_ab1[i] = ae_b1[i]; }
  for (int i = tid; i < 64*32;  i += 256) { s_mw2[i] = me_w2[i]; s_aw2[i] = ae_w2[i]; }
  for (int i = tid; i < 64;     i += 256) { s_mb2[i] = me_b2[i]; s_ab2[i] = ae_b2[i];
                                            s_ob[i] = ob[i]; s_g[i] = ln_g[i]; s_bb[i] = ln_b[i]; }
  for (int i = tid; i < 32*6;   i += 256) s_aw1[i] = ae_w1[i];
  for (int i = tid; i < 192*64; i += 256) s_inw[i] = in_w[i];
  for (int i = tid; i < 192;    i += 256) s_inb[i] = in_b[i];
  for (int i = tid; i < 64*64;  i += 256) s_ow[i]  = ow[i];
  __syncthreads();

  const int p = blockIdx.x * 256 + tid;
  const float* o = obs + (size_t)p * 12;

  float m1e[64], m2e[64], ace[64];
  // motion encoder (shared weights) for m1 and m2
  {
    float t1[32], t2[32];
    #pragma unroll 4
    for (int i = 0; i < 32; ++i) {
      const float* w = s_mw1 + i * 3;
      t1[i] = leaky(s_mb1[i] + w[0]*o[0] + w[1]*o[1] + w[2]*o[2]);
      t2[i] = leaky(s_mb1[i] + w[0]*o[3] + w[1]*o[4] + w[2]*o[5]);
    }
    for (int q = 0; q < 64; ++q) {
      const float* w = s_mw2 + q * 32;
      float a1 = s_mb2[q], a2 = s_mb2[q];
      #pragma unroll 8
      for (int i = 0; i < 32; ++i) { a1 += w[i]*t1[i]; a2 += w[i]*t2[i]; }
      m1e[q] = a1; m2e[q] = a2;
    }
  }
  // action encoder
  {
    float t[32];
    #pragma unroll 4
    for (int i = 0; i < 32; ++i) {
      const float* w = s_aw1 + i * 6;
      float a = s_ab1[i];
      #pragma unroll
      for (int j = 0; j < 6; ++j) a += w[j]*o[6 + j];
      t[i] = leaky(a);
    }
    for (int q = 0; q < 64; ++q) {
      const float* w = s_aw2 + q * 32;
      float a = s_ab2[q];
      #pragma unroll 8
      for (int i = 0; i < 32; ++i) a += w[i]*t[i];
      ace[q] = a;
    }
  }

  // attention over 2 keys per head
  float ctx[64];
  const float scale = 0.25f;  // 1/sqrt(16)
  for (int h = 0; h < NHm; ++h) {
    const int base = h * HDm;
    float qv[16], k1v[16], k2v[16], v1v[16], v2v[16];
    for (int d = 0; d < HDm; ++d) {
      const float* wq = s_inw + (base + d) * 64;
      const float* wk = s_inw + (64  + base + d) * 64;
      const float* wv = s_inw + (128 + base + d) * 64;
      float aq = s_inb[base + d];
      float a1 = s_inb[64 + base + d],  a2 = a1;
      float b1 = s_inb[128 + base + d], b2 = b1;
      #pragma unroll 8
      for (int c = 0; c < 64; ++c) {
        aq += wq[c]*ace[c];
        a1 += wk[c]*m1e[c]; a2 += wk[c]*m2e[c];
        b1 += wv[c]*m1e[c]; b2 += wv[c]*m2e[c];
      }
      qv[d]=aq; k1v[d]=a1; k2v[d]=a2; v1v[d]=b1; v2v[d]=b2;
    }
    float s1 = 0.f, s2 = 0.f;
    #pragma unroll
    for (int d = 0; d < HDm; ++d) { s1 += qv[d]*k1v[d]; s2 += qv[d]*k2v[d]; }
    s1 *= scale; s2 *= scale;
    float mx = fmaxf(s1, s2);
    float e1 = __expf(s1 - mx), e2 = __expf(s2 - mx);
    float inv = 1.f / (e1 + e2);
    #pragma unroll
    for (int d = 0; d < HDm; ++d) ctx[base + d] = (e1*v1v[d] + e2*v2v[d]) * inv;
  }

  // out_proj + residual + layernorm
  float x[64]; float mu = 0.f;
  for (int q = 0; q < 64; ++q) {
    const float* w = s_ow + q * 64;
    float a = s_ob[q] + ace[q];
    #pragma unroll 8
    for (int c = 0; c < 64; ++c) a += w[c]*ctx[c];
    x[q] = a; mu += a;
  }
  mu *= (1.f / 64.f);
  float var = 0.f;
  #pragma unroll 8
  for (int q = 0; q < 64; ++q) { float d = x[q] - mu; var += d*d; }
  var *= (1.f / 64.f);
  const float rstd = rsqrtf(var + LN_EPS);
  _Float16* dst = xn_out + (size_t)p * 64;
  #pragma unroll 8
  for (int q = 0; q < 64; ++q)
    dst[q] = (_Float16)(((x[q] - mu) * rstd) * s_g[q] + s_bb[q]);
}

// ============================================================================
// Kernel 2: xp = xn @ W_ih.T + b_ih   (M = BS, N = 384, K = 64)  -- WMMA
// 4 waves/block, each wave owns one 16-row M tile; results staged through a
// per-wave LDS tile so the 96 MB xp stream goes out as coalesced b128 stores.
// ============================================================================
__global__ __launch_bounds__(128) void k2_xproj(
    const _Float16* __restrict__ xn, const float* __restrict__ w_ih,
    const float* __restrict__ b_ih,  _Float16* __restrict__ xp)
{
  __shared__ __align__(16) _Float16 sW[384 * 64];        // 48 KB, W_ih in f16
  __shared__ __align__(16) _Float16 sOut[4][16 * 384];   // 48 KB, per-wave out tile
  __shared__ float sB[384];
  const int tid = threadIdx.x;
  for (int i = tid; i < 384*64; i += 128) sW[i] = (_Float16)w_ih[i];
  for (int i = tid; i < 384;    i += 128) sB[i] = b_ih[i];
  __syncthreads();

  const int lane = tid & 31, wave = tid >> 5;
  const int col = lane & 15, rb = (lane & 16) ? 8 : 0;
  const size_t mtile = (size_t)blockIdx.x * 4 + wave;
  const _Float16* A = xn + mtile * 16 * 64;

  for (int nt = 0; nt < 24; ++nt) {
    const int nb = nt * 16;
    const float bv = sB[nb + col];
    v8f c;
    #pragma unroll
    for (int i = 0; i < 8; ++i) c[i] = bv;
    #pragma unroll
    for (int ks = 0; ks < 2; ++ks) {
      v16h a = load_tileA(A, 64, ks * 32, lane);
      v16h b = load_tileB(sW, 64, nb, ks * 32, lane);
      c = wmma16(a, b, c);
    }
    #pragma unroll
    for (int i = 0; i < 8; ++i)
      sOut[wave][(rb + i) * 384 + nb + col] = (_Float16)c[i];
  }
  __syncthreads();

  // coalesced b128 write-out of the 16x384 tile
  _Float16* O = xp + mtile * 16 * 384;
  for (int c2 = lane; c2 < (16 * 384) / 8; c2 += 32)
    *(v8h*)(O + (size_t)c2 * 8) = *(const v8h*)(&sOut[wave][c2 * 8]);
}

// ============================================================================
// Kernel 3: GRU scan. 8 blocks x 16 batch rows. Wave w owns the SAME 16-col
// block (n0 = w*16) of all three gates, so r/z/g for a (row,col) live in one
// lane's accumulators -> gates fully in registers, no hp LDS round-trip.
// W_hh B-tiles + biases hoisted to registers; h kept in f32 registers with a
// double-buffered f16 shadow in LDS for the WMMA A operand (1 barrier/step).
// ============================================================================
__global__ __launch_bounds__(256) void k3_gru(
    const _Float16* __restrict__ xp, const float* __restrict__ w_hh,
    const float* __restrict__ b_hh,  const float* __restrict__ h0,
    _Float16* __restrict__ gru_out,  float* __restrict__ h_last)
{
  __shared__ __align__(16) _Float16 sH[2][16 * 128];   // ping-pong f16 h (8 KB)

  const int tid = threadIdx.x, lane = tid & 31, wave = tid >> 5;
  const int bg0 = blockIdx.x * 16;
  const int col = lane & 15, rb = (lane & 16) ? 8 : 0;
  const int n0 = wave * 16;   // this wave's column block within each gate

  // hoist W_hh B tiles for gates r/z/g (12 x v16h = 96 VGPRs) and biases
  v16h Br[4], Bz[4], Bg[4];
  #pragma unroll
  for (int ks = 0; ks < 4; ++ks) {
    Br[ks] = load_tileB_f32(w_hh, 128, n0,        ks * 32, lane);
    Bz[ks] = load_tileB_f32(w_hh, 128, 128 + n0,  ks * 32, lane);
    Bg[ks] = load_tileB_f32(w_hh, 128, 256 + n0,  ks * 32, lane);
  }
  const float br = b_hh[n0 + col];
  const float bz = b_hh[128 + n0 + col];
  const float bg = b_hh[256 + n0 + col];

  // h held in registers at this lane's C-layout positions (rows rb..rb+7, col n0+col)
  float h[8];
  #pragma unroll
  for (int i = 0; i < 8; ++i) {
    h[i] = h0[(size_t)(bg0 + rb + i) * 128 + n0 + col];
    sH[0][(rb + i) * 128 + n0 + col] = (_Float16)h[i];
  }
  __syncthreads();

  for (int t = 0; t < Sm; ++t) {
    const int cur = t & 1, nxt = cur ^ 1;

    // preload this step's xp gate inputs (independent of h) + prefetch t+1
    float xr[8], xz[8], xg[8];
    #pragma unroll
    for (int i = 0; i < 8; ++i) {
      const size_t xb = ((size_t)(bg0 + rb + i) * Sm + t) * 384 + n0 + col;
      xr[i] = (float)xp[xb];
      xz[i] = (float)xp[xb + 128];
      xg[i] = (float)xp[xb + 256];
      if (t + 1 < Sm) __builtin_prefetch(xp + xb + 384, 0, 3);
    }

    // hp tiles: three independent 4-deep WMMA chains sharing the A tile
    v8f cr, cz, cg;
    #pragma unroll
    for (int i = 0; i < 8; ++i) { cr[i] = br; cz[i] = bz; cg[i] = bg; }
    #pragma unroll
    for (int ks = 0; ks < 4; ++ks) {
      v16h a = load_tileA(sH[cur], 128, ks * 32, lane);
      cr = wmma16(a, Br[ks], cr);
      cz = wmma16(a, Bz[ks], cz);
      cg = wmma16(a, Bg[ks], cg);
    }

    // fused gates entirely in registers
    #pragma unroll
    for (int i = 0; i < 8; ++i) {
      const float r  = 1.f / (1.f + __expf(-(xr[i] + cr[i])));
      const float z  = 1.f / (1.f + __expf(-(xz[i] + cz[i])));
      const float g  = tanhf(xg[i] + r * cg[i]);
      const float hn = (1.f - z) * g + z * h[i];
      h[i] = hn;
      sH[nxt][(rb + i) * 128 + n0 + col] = (_Float16)hn;
      gru_out[((size_t)(bg0 + rb + i) * Sm + t) * 128 + n0 + col] = (_Float16)hn;
    }
    __syncthreads();
  }

  #pragma unroll
  for (int i = 0; i < 8; ++i)
    h_last[(size_t)(bg0 + rb + i) * 128 + n0 + col] = h[i];
}

// ============================================================================
// Kernel 4: MLP 128 -> 128 -> 128 -> 1 over all positions.  WMMA both layers,
// weights in LDS, per-wave f16 activation scratch (double-buffered).
// ============================================================================
__global__ __launch_bounds__(256) void k4_mlp(
    const _Float16* __restrict__ go,
    const float* __restrict__ w0, const float* __restrict__ b0,
    const float* __restrict__ w1, const float* __restrict__ b1,
    const float* __restrict__ ow, const float* __restrict__ ob,
    float* __restrict__ value)
{
  __shared__ __align__(16) _Float16 sW0[128 * 128];      // 32 KB
  __shared__ __align__(16) _Float16 sW1[128 * 128];      // 32 KB
  __shared__ __align__(16) _Float16 sF [8][16 * 128];    // 32 KB
  __shared__ __align__(16) _Float16 sF2[8][16 * 128];    // 32 KB
  __shared__ float sB0[128], sB1[128], sOW[128];

  const int tid = threadIdx.x, lane = tid & 31, wave = tid >> 5;
  for (int i = tid; i < 128*128; i += 256) { sW0[i] = (_Float16)w0[i]; sW1[i] = (_Float16)w1[i]; }
  for (int i = tid; i < 128;     i += 256) { sB0[i] = b0[i]; sB1[i] = b1[i]; sOW[i] = ow[i]; }
  __syncthreads();

  const int col = lane & 15, rb = (lane & 16) ? 8 : 0;
  const size_t m0 = ((size_t)blockIdx.x * 8 + wave) * 16;
  const _Float16* A0 = go + m0 * 128;

  // layer 0: f = leaky(go @ W0.T + b0)
  for (int nt = 0; nt < 8; ++nt) {
    const int nb = nt * 16;
    const float bv = sB0[nb + col];
    v8f c;
    #pragma unroll
    for (int i = 0; i < 8; ++i) c[i] = bv;
    #pragma unroll
    for (int ks = 0; ks < 4; ++ks) {
      v16h a = load_tileA(A0, 128, ks * 32, lane);
      v16h b = load_tileB(sW0, 128, nb, ks * 32, lane);
      c = wmma16(a, b, c);
    }
    #pragma unroll
    for (int i = 0; i < 8; ++i)
      sF[wave][(rb + i) * 128 + nb + col] = (_Float16)leaky(c[i]);
  }
  __syncthreads();

  // layer 1: f = leaky(f @ W1.T + b1)
  for (int nt = 0; nt < 8; ++nt) {
    const int nb = nt * 16;
    const float bv = sB1[nb + col];
    v8f c;
    #pragma unroll
    for (int i = 0; i < 8; ++i) c[i] = bv;
    #pragma unroll
    for (int ks = 0; ks < 4; ++ks) {
      v16h a = load_tileA(sF[wave], 128, ks * 32, lane);
      v16h b = load_tileB(sW1, 128, nb, ks * 32, lane);
      c = wmma16(a, b, c);
    }
    #pragma unroll
    for (int i = 0; i < 8; ++i)
      sF2[wave][(rb + i) * 128 + nb + col] = (_Float16)leaky(c[i]);
  }
  __syncthreads();

  // output head: value = f @ ow.T + ob  (dot of 128 per row; lanes 0-15 = rows)
  if (lane < 16) {
    float a = ob[0];
    const _Float16* fr = &sF2[wave][lane * 128];
    #pragma unroll 8
    for (int c2 = 0; c2 < 128; ++c2) a += (float)fr[c2] * sOW[c2];
    value[m0 + lane] = a;
  }
}

// ============================================================================
extern "C" void kernel_launch(void* const* d_in, const int* in_sizes, int n_in,
                              void* d_out, int out_size, void* d_ws, size_t ws_size,
                              hipStream_t stream)
{
  const float* obs      = (const float*)d_in[0];
  const float* h0       = (const float*)d_in[1];
  const float* me_w1    = (const float*)d_in[2];
  const float* me_b1    = (const float*)d_in[3];
  const float* me_w2    = (const float*)d_in[4];
  const float* me_b2    = (const float*)d_in[5];
  const float* ae_w1    = (const float*)d_in[6];
  const float* ae_b1    = (const float*)d_in[7];
  const float* ae_w2    = (const float*)d_in[8];
  const float* ae_b2    = (const float*)d_in[9];
  const float* in_w     = (const float*)d_in[10];
  const float* in_b     = (const float*)d_in[11];
  const float* out_w    = (const float*)d_in[12];
  const float* out_b    = (const float*)d_in[13];
  const float* ln_g     = (const float*)d_in[14];
  const float* ln_b     = (const float*)d_in[15];
  const float* gru_w_ih = (const float*)d_in[16];
  const float* gru_w_hh = (const float*)d_in[17];
  const float* gru_b_ih = (const float*)d_in[18];
  const float* gru_b_hh = (const float*)d_in[19];
  const float* mlp_w0   = (const float*)d_in[20];
  const float* mlp_b0   = (const float*)d_in[21];
  const float* mlp_w1   = (const float*)d_in[22];
  const float* mlp_b1   = (const float*)d_in[23];
  const float* o_w      = (const float*)d_in[24];
  const float* o_b      = (const float*)d_in[25];

  char* ws = (char*)d_ws;
  _Float16* ws_xn = (_Float16*)ws;                                        // BS*64  f16 = 16 MB
  _Float16* ws_xp = (_Float16*)(ws + (size_t)BS * 64 * 2);                // BS*384 f16 = 96 MB
  _Float16* ws_go = (_Float16*)(ws + (size_t)BS * 64 * 2 + (size_t)BS * 384 * 2); // BS*128 f16 = 32 MB

  float* value  = (float*)d_out;       // B*S
  float* h_last = value + BS;          // B*H

  k1_encode_attn_ln<<<BS / 256, 256, 0, stream>>>(
      obs, me_w1, me_b1, me_w2, me_b2, ae_w1, ae_b1, ae_w2, ae_b2,
      in_w, in_b, out_w, out_b, ln_g, ln_b, ws_xn);

  k2_xproj<<<BS / (16 * 4), 128, 0, stream>>>(ws_xn, gru_w_ih, gru_b_ih, ws_xp);

  k3_gru<<<Bm / 16, 256, 0, stream>>>(ws_xp, gru_w_hh, gru_b_hh, h0, ws_go, h_last);

  k4_mlp<<<BS / (16 * 8), 256, 0, stream>>>(ws_go, mlp_w0, mlp_b0, mlp_w1, mlp_b1,
                                            o_w, o_b, value);
}
